// GATEncoder_16389595202065
// MI455X (gfx1250) — matrix operand order repaired and checked by reading, not compiled, not run on hardware
//
#include <hip/hip_runtime.h>

// GAT encoder for MI455X (gfx1250, wave32, WMMA + async global->LDS staging).
// att@Wh done flash-style with v_wmma_f32_16x16x32_f16; P tiles regenerated
// on the fly (branchless) from rank-1 scores + adj mask; 1/denom folded into
// the f32 accumulator at the end. adj (64MB) is L2-resident (192MB L2).

#define N_NODES 4096
#define DIM     128
#define HEADS   8
#define HDIM    16
#define LAYERS  2
#define ALPHA_S 0.2f
#define LN_EPS  1e-5f

#define USE_ASYNC_STAGE 1   // gfx1250 global_load_async_to_lds_* staging

typedef __attribute__((ext_vector_type(16))) _Float16 v16h;
typedef __attribute__((ext_vector_type(8)))  _Float16 v8h;
typedef __attribute__((ext_vector_type(2)))  _Float16 h2v;
typedef __attribute__((ext_vector_type(8)))  float    v8f;

__device__ __forceinline__ float warp_max(float v) {
#pragma unroll
  for (int o = 16; o > 0; o >>= 1) v = fmaxf(v, __shfl_xor(v, o, 32));
  return v;
}
__device__ __forceinline__ float warp_sum(float v) {
#pragma unroll
  for (int o = 16; o > 0; o >>= 1) v += __shfl_xor(v, o, 32);
  return v;
}
__device__ __forceinline__ float elu_f(float x) {
  return x > 0.f ? x : (__expf(x) - 1.f);
}
__device__ __forceinline__ h2v pk2(float x, float y) {
  auto r = __builtin_amdgcn_cvt_pkrtz(x, y);   // v_cvt_pk_f16_f32
  return *(h2v*)&r;
}

// x[n,d] = nf[n,0]*W[0,d] + nf[n,1]*W[1,d] + b[d]
__global__ void gat_embed(const float* __restrict__ nf, const float* __restrict__ W,
                          const float* __restrict__ b, float* __restrict__ x) {
  int id = blockIdx.x * blockDim.x + threadIdx.x;   // N*DIM threads
  int n = id >> 7, d = id & 127;
  x[id] = fmaf(nf[2 * n], W[d], fmaf(nf[2 * n + 1], W[DIM + d], b[d]));
}

// Wh[h,n,k] = sum_d x[n,d] * Wl[h,d,k]; also transposed f16 copy WhT[h*16+k][n]
__global__ void gat_proj(const float* __restrict__ x, const float* __restrict__ Wl,
                         float* __restrict__ Wh, _Float16* __restrict__ WhT) {
  __shared__ float xs[2 * DIM];
  int t = threadIdx.x;
  int n0 = blockIdx.x * 2;
  xs[t] = x[n0 * DIM + t];
  __syncthreads();
  int nloc = t >> 7;
  int col = t & 127;
  int h = col >> 4, k = col & 15;
  const float* wcol = Wl + (h * DIM) * HDIM + k;   // stride HDIM over d
  const float* xr = xs + nloc * DIM;
  float s = 0.f;
#pragma unroll 8
  for (int d = 0; d < DIM; ++d) s = fmaf(xr[d], wcol[d * HDIM], s);
  int n = n0 + nloc;
  Wh[(h * N_NODES + n) * HDIM + k] = s;
  WhT[(h * HDIM + k) * N_NODES + n] = (_Float16)s;
}

// f1[h,n] = Wh[h,n,:]·a[h,:16]; f2[h,n] = Wh[h,n,:]·a[h,16:]
__global__ void gat_f12(const float* __restrict__ Wh, const float* __restrict__ al,
                        float* __restrict__ f1, float* __restrict__ f2) {
  int id = blockIdx.x * blockDim.x + threadIdx.x;   // HEADS*N threads
  int h = id >> 12, n = id & (N_NODES - 1);
  const float* w = Wh + (h * N_NODES + n) * HDIM;
  const float* a = al + h * 2 * HDIM;
  float s1 = 0.f, s2 = 0.f;
#pragma unroll
  for (int k = 0; k < HDIM; ++k) {
    s1 = fmaf(w[k], a[k], s1);
    s2 = fmaf(w[k], a[HDIM + k], s2);
  }
  f1[id] = s1;
  f2[id] = s2;
}

// Pass 1: per (h,n) rowmax = lrelu(f1 + masked_max(f2)) (lrelu monotone), then
// denom. One wave per row n, all 8 heads per wave. Fully branchless selects.
__global__ void gat_stats(const int* __restrict__ adj, const float* __restrict__ f1,
                          const float* __restrict__ f2, float* __restrict__ rowmax,
                          float* __restrict__ invden) {
  int lane = threadIdx.x & 31;
  int wave = threadIdx.x >> 5;
  int n = blockIdx.x * 8 + wave;
  const int4* arow = (const int4*)(adj + (long)n * N_NODES);

  float f1h[HEADS], mx[HEADS];
#pragma unroll
  for (int h = 0; h < HEADS; ++h) {
    f1h[h] = f1[h * N_NODES + n];
    mx[h] = -INFINITY;
  }
  // Phase A: masked max of f2 per head (cndmask, no branches)
  for (int it = lane; it < N_NODES / 4; it += 32) {
    int4 a4 = arow[it];
    int m = it * 4;
#pragma unroll
    for (int h = 0; h < HEADS; ++h) {
      float4 g = *(const float4*)(f2 + h * N_NODES + m);
      mx[h] = fmaxf(mx[h], a4.x ? g.x : -INFINITY);
      mx[h] = fmaxf(mx[h], a4.y ? g.y : -INFINITY);
      mx[h] = fmaxf(mx[h], a4.z ? g.z : -INFINITY);
      mx[h] = fmaxf(mx[h], a4.w ? g.w : -INFINITY);
    }
  }
  float rmx[HEADS];
#pragma unroll
  for (int h = 0; h < HEADS; ++h) {
    float m2 = warp_max(mx[h]);
    float tv = f1h[h] + m2;
    float e = tv > 0.f ? tv : ALPHA_S * tv;
    rmx[h] = (e > -1e30f) ? e : 0.f;   // degenerate no-neighbor guard
  }
  // Phase B: denominator; exp runs unconditionally, mask selects the argument
  float sm[HEADS];
#pragma unroll
  for (int h = 0; h < HEADS; ++h) sm[h] = 0.f;
  for (int it = lane; it < N_NODES / 4; it += 32) {
    int4 a4 = arow[it];
    int m = it * 4;
#pragma unroll
    for (int h = 0; h < HEADS; ++h) {
      float4 g = *(const float4*)(f2 + h * N_NODES + m);
      float t0 = f1h[h] + g.x, t1 = f1h[h] + g.y, t2 = f1h[h] + g.z, t3 = f1h[h] + g.w;
      float e0 = t0 > 0.f ? t0 : ALPHA_S * t0;
      float e1 = t1 > 0.f ? t1 : ALPHA_S * t1;
      float e2 = t2 > 0.f ? t2 : ALPHA_S * t2;
      float e3 = t3 > 0.f ? t3 : ALPHA_S * t3;
      sm[h] += __expf(a4.x ? (e0 - rmx[h]) : -1e4f);
      sm[h] += __expf(a4.y ? (e1 - rmx[h]) : -1e4f);
      sm[h] += __expf(a4.z ? (e2 - rmx[h]) : -1e4f);
      sm[h] += __expf(a4.w ? (e3 - rmx[h]) : -1e4f);
    }
  }
#pragma unroll
  for (int h = 0; h < HEADS; ++h) {
    float s = warp_sum(sm[h]);
    if (lane == 0) {
      rowmax[h * N_NODES + n] = rmx[h];
      invden[h * N_NODES + n] = 1.f / fmaxf(s, 1e-30f);
    }
  }
}

// Pass 2: h_out = elu((P @ Wh) * invden). One workgroup per 16-row block,
// wave w == head w. Double-buffered LDS tiles staged with async global->LDS
// copies (ASYNCcnt), one barrier per 32-wide K block, WMMA accumulate.
__global__ void __launch_bounds__(256) gat_attn(
    const int* __restrict__ adj, const float* __restrict__ f1,
    const float* __restrict__ f2, const float* __restrict__ rowmax,
    const float* __restrict__ invden, const _Float16* __restrict__ WhT,
    float* __restrict__ xo) {
  // stride 36: conflict-free (r*36 mod 64 distinct for r=0..15) and 16B aligned
  __shared__ __align__(16) int   adjs[2][16][36];
  __shared__ __align__(16) float f2s[2][HEADS][32];

  int t = threadIdx.x;
  int lane = t & 31;
  int h = t >> 5;
  int n0 = blockIdx.x * 16;
  int M = lane & 15;            // A row (and B column / output feature index)
  int hi = lane >> 4;
  int kbase = hi ? 8 : 0;       // 16-bit A layout: lane<16 K∈{0..7,16..23}, else {8..15,24..31}

  float f1n = f1[h * N_NODES + n0 + M];
  float rmx = rowmax[h * N_NODES + n0 + M];

  // staging geometry: each thread moves one b64 of adj + one b32 of f2
  int rA = t >> 4;              // adj tile row 0..15
  int cA = (t << 1) & 31;       // adj tile col (even)
  int hF = t >> 5, cF = t & 31; // f2 tile coords
  unsigned gAoff = ((unsigned)(n0 + rA) * N_NODES + cA) * 4u;  // + m0*4
  unsigned gFoff = ((unsigned)hF * N_NODES + cF) * 4u;         // + m0*4
  unsigned ldsA[2], ldsF[2];
#pragma unroll
  for (int b = 0; b < 2; ++b) {
    ldsA[b] = (unsigned)(uintptr_t)&adjs[b][rA][cA];
    ldsF[b] = (unsigned)(uintptr_t)&f2s[b][hF][cF];
  }

#if USE_ASYNC_STAGE
#define STAGE(IT, B)                                                          \
  do {                                                                        \
    unsigned moff = (unsigned)(IT) * 32u * 4u;                                \
    asm volatile("global_load_async_to_lds_b64 %0, %1, %2 offset:0"           \
                 :: "v"(ldsA[B]), "v"(gAoff + moff), "s"(adj) : "memory");    \
    asm volatile("global_load_async_to_lds_b32 %0, %1, %2 offset:0"           \
                 :: "v"(ldsF[B]), "v"(gFoff + moff), "s"(f2) : "memory");     \
  } while (0)
#define WAIT_STAGE() asm volatile("s_wait_asynccnt 0x0" ::: "memory")
#else
  int2 avs; float fvs;
#define STAGE(IT, B)                                                          \
  do {                                                                        \
    unsigned moff = (unsigned)(IT) * 32u;                                     \
    avs = *(const int2*)(adj + (unsigned)(n0 + rA) * N_NODES + cA + moff);    \
    fvs = f2[(unsigned)hF * N_NODES + cF + moff];                             \
    *(int2*)((char*)0 + (uintptr_t)ldsA[B]) = avs;                            \
    *(float*)((char*)0 + (uintptr_t)ldsF[B]) = fvs;                           \
  } while (0)
#define WAIT_STAGE() do {} while (0)
#endif

  v8f acc = {};
  constexpr int NIT = N_NODES / 32;

  STAGE(0, 0);
  WAIT_STAGE();
  __syncthreads();

  for (int it = 0; it < NIT; ++it) {
    int b = it & 1;
    if (it + 1 < NIT) STAGE(it + 1, b ^ 1);   // overlaps with compute below

    // ---- build A tile (branchless): vector LDS reads, exp unconditionally
    const int*   ar = &adjs[b][M][0];
    const float* gr = &f2s[b][h][0];
    int4   m0v = *(const int4*)(ar + kbase);
    int4   m1v = *(const int4*)(ar + kbase + 4);
    int4   m2v = *(const int4*)(ar + kbase + 16);
    int4   m3v = *(const int4*)(ar + kbase + 20);
    float4 g0 = *(const float4*)(gr + kbase);
    float4 g1 = *(const float4*)(gr + kbase + 4);
    float4 g2 = *(const float4*)(gr + kbase + 16);
    float4 g3 = *(const float4*)(gr + kbase + 20);

#define PE(G, MK)                                                             \
  ({ float _tv = f1n + (G);                                                   \
     float _e = _tv > 0.f ? _tv : ALPHA_S * _tv;                              \
     __expf((MK) ? (_e - rmx) : -1e4f); })

    union { v16h v; h2v p[8]; } A;
    A.p[0] = pk2(PE(g0.x, m0v.x), PE(g0.y, m0v.y));
    A.p[1] = pk2(PE(g0.z, m0v.z), PE(g0.w, m0v.w));
    A.p[2] = pk2(PE(g1.x, m1v.x), PE(g1.y, m1v.y));
    A.p[3] = pk2(PE(g1.z, m1v.z), PE(g1.w, m1v.w));
    A.p[4] = pk2(PE(g2.x, m2v.x), PE(g2.y, m2v.y));
    A.p[5] = pk2(PE(g2.z, m2v.z), PE(g2.w, m2v.w));
    A.p[6] = pk2(PE(g3.x, m3v.x), PE(g3.y, m3v.y));
    A.p[7] = pk2(PE(g3.z, m3v.z), PE(g3.w, m3v.w));
#undef PE

    // ---- B tile: lanes<16 hold K={0..7,16..23} of column N=lane,
    // lanes>=16 hold K={8..15,24..31} — two b128 loads from transposed Wh.
    const v8h* bp =
        (const v8h*)(WhT + ((h * HDIM + M) * N_NODES + it * 32 + hi * 8));
    v8h b0 = bp[0], b1 = bp[2];
    v16h B;
#pragma unroll
    for (int i = 0; i < 8; ++i) { B[i] = b0[i]; B[i + 8] = b1[i]; }

    acc = __builtin_amdgcn_wmma_f32_16x16x32_f16(
        /*neg_a=*/false, A.v, /*neg_b=*/false, B,
        /*c_mod=*/(short)0, acc, /*reuse_a=*/false, /*reuse_b=*/false);

    WAIT_STAGE();
    __syncthreads();
  }
#undef STAGE
#undef WAIT_STAGE

  // C/D layout: VGPR r holds row M=r (lanes<16) or r+8 (lanes>=16), col N=lane&15.
  int col = h * HDIM + M;
#pragma unroll
  for (int r = 0; r < 8; ++r) {
    int row = n0 + r + 8 * hi;
    float v = acc[r] * invden[h * N_NODES + row];
    v = v > 0.f ? v : (__expf(v) - 1.f);      // ELU before concat/LN (as reference)
    xo[(long)row * DIM + col] = v;
  }
}

// LayerNorm over D=128 per row + ELU. One wave per row, 4 floats per lane.
__global__ void gat_ln(const float* __restrict__ xin, const float* __restrict__ gamma,
                       const float* __restrict__ beta, float* __restrict__ xout) {
  int lane = threadIdx.x & 31;
  int wave = threadIdx.x >> 5;
  int n = blockIdx.x * 8 + wave;
  int c = lane * 4;
  float4 v = *(const float4*)(xin + (long)n * DIM + c);
  float s = v.x + v.y + v.z + v.w;
  float sq = v.x * v.x + v.y * v.y + v.z * v.z + v.w * v.w;
  s = warp_sum(s);
  sq = warp_sum(sq);
  float mu = s * (1.f / DIM);
  float var = sq * (1.f / DIM) - mu * mu;
  float rinv = rsqrtf(var + LN_EPS);
  float4 o;
  o.x = elu_f(gamma[c + 0] * (v.x - mu) * rinv + beta[c + 0]);
  o.y = elu_f(gamma[c + 1] * (v.y - mu) * rinv + beta[c + 1]);
  o.z = elu_f(gamma[c + 2] * (v.z - mu) * rinv + beta[c + 2]);
  o.w = elu_f(gamma[c + 3] * (v.w - mu) * rinv + beta[c + 3]);
  *(float4*)(xout + (long)n * DIM + c) = o;
}

extern "C" void kernel_launch(void* const* d_in, const int* in_sizes, int n_in,
                              void* d_out, int out_size, void* d_ws, size_t ws_size,
                              hipStream_t stream) {
  const float* nf   = (const float*)d_in[0];
  const int*   adj  = (const int*)d_in[1];
  const float* Wemb = (const float*)d_in[2];
  const float* bemb = (const float*)d_in[3];
  const float* Watt = (const float*)d_in[4];   // [L,H,D,HD]
  const float* aatt = (const float*)d_in[5];   // [L,H,2*HD]
  const float* lng  = (const float*)d_in[6];   // [L,D]
  const float* lnb  = (const float*)d_in[7];   // [L,D]

  char* ws = (char*)d_ws;
  float*    x    = (float*)(ws + (0ull << 20));            // 2 MB
  float*    xn   = (float*)(ws + (2ull << 20));            // 2 MB
  float*    Wh   = (float*)(ws + (4ull << 20));            // 2 MB
  _Float16* WhT  = (_Float16*)(ws + (6ull << 20));         // 1 MB
  float*    f1   = (float*)(ws + (7ull << 20));            // 128 KB
  float*    f2   = (float*)(ws + (7ull << 20) + (128u << 10));
  float*    rmx  = (float*)(ws + (7ull << 20) + (256u << 10));
  float*    idn  = (float*)(ws + (7ull << 20) + (384u << 10));

  gat_embed<<<N_NODES * DIM / 256, 256, 0, stream>>>(nf, Wemb, bemb, x);
  for (int l = 0; l < LAYERS; ++l) {
    const float* Wl = Watt + (size_t)l * HEADS * DIM * HDIM;
    const float* al = aatt + (size_t)l * HEADS * 2 * HDIM;
    gat_proj<<<N_NODES / 2, 256, 0, stream>>>(x, Wl, Wh, WhT);
    gat_f12<<<HEADS * N_NODES / 256, 256, 0, stream>>>(Wh, al, f1, f2);
    gat_stats<<<N_NODES / 8, 256, 0, stream>>>(adj, f1, f2, rmx, idn);
    gat_attn<<<N_NODES / 16, 256, 0, stream>>>(adj, f1, f2, rmx, idn, WhT, xn);
    float* dst = (l == LAYERS - 1) ? (float*)d_out : x;
    gat_ln<<<N_NODES / 8, 256, 0, stream>>>(xn, lng + l * DIM, lnb + l * DIM, dst);
  }
}
